// SparseAttention_40948218200689
// MI455X (gfx1250) — matrix operand order repaired and checked by reading, not compiled, not run on hardware
//
#include <hip/hip_runtime.h>
#include <hip/hip_bf16.h>

// ---------------------------------------------------------------------------
// Sliding-window causal attention block for MI455X (gfx1250, wave32, WMMA).
//   1) qkv = x @ W_qkv + b_qkv              (bf16 WMMA, bf16 workspace out)
//   2) windowed causal attention (win=16)    (WMMA scores + WMMA P·V)
//   3) out = (attn * gate) @ W_proj + b_proj (bf16 WMMA, fp32 out)
// Memory-bound problem: intermediates kept in bf16 to halve HBM traffic.
// ---------------------------------------------------------------------------

typedef __bf16 v8bf  __attribute__((ext_vector_type(8)));
typedef __bf16 v16bf __attribute__((ext_vector_type(16)));
typedef float  v8f   __attribute__((ext_vector_type(8)));

static constexpr int kB   = 512;
static constexpr int kT   = 128;
static constexpr int kC   = 1024;
static constexpr int kH   = 8;
static constexpr int kD   = 128;         // head dim
static constexpr int kN1  = 3 * kC;      // 3072
static constexpr int kWin = 16;

__device__ __forceinline__ v8bf ld8(const __bf16* p) {
    return *reinterpret_cast<const v8bf*>(p);
}
__device__ __forceinline__ v16bf join16(v8bf lo, v8bf hi) {
    return __builtin_shufflevector(lo, hi, 0,1,2,3,4,5,6,7,8,9,10,11,12,13,14,15);
}
// A-matrix 16x32 bf16 fragment (ISA 7.12.2): lane holds row (lane&15); the 16
// elements are two contiguous 8-runs of K at half*8 and 16+half*8.
__device__ __forceinline__ v16bf loadA(const __bf16* row, int half) {
    return join16(ld8(row + half * 8), ld8(row + 16 + half * 8));
}
// B-matrix 32x16 bf16 fragment: lane holds column (lane&15); the 16 elements
// are two contiguous 8-runs of K at half*8 and 16+half*8 (matches A).
__device__ __forceinline__ v16bf loadB(const __bf16* row, int half) {
    return join16(ld8(row + half * 8), ld8(row + 16 + half * 8));
}
__device__ __forceinline__ v8f wmma_bf16(v16bf a, v16bf b, v8f c) {
    return __builtin_amdgcn_wmma_f32_16x16x32_bf16(false, a, false, b,
                                                   (short)0, c, false, false);
}

// ---------------------------------------------------------------------------
// GEMM 1: qkv[M=65536, N=3072] = x[M,K=1024] @ W[K,N] + bias  -> bf16
// Block: 256 threads (8 waves, 4x2), tile 128(M) x 64(N), K-chunk 32.
// ---------------------------------------------------------------------------
__global__ __launch_bounds__(256) void qkv_gemm_kernel(
    const float* __restrict__ x, const float* __restrict__ W,
    const float* __restrict__ bias, __bf16* __restrict__ out)
{
    constexpr int K = kC, N = kN1, TM = 128, TN = 64, TK = 32;
    __shared__ __align__(16) __bf16 As[TM][TK];
    __shared__ __align__(16) __bf16 BsT[TN][TK];

    const int tid  = threadIdx.x;
    const int wave = tid >> 5, lane = tid & 31;
    const int lo16 = lane & 15, half = lane >> 4;
    const int waveM = wave >> 1, waveN = wave & 1;
    const size_t mBase = (size_t)blockIdx.y * TM;
    const int    nBase = blockIdx.x * TN;

    v8f acc[2][2] = {};

    for (int kc = 0; kc < K; kc += TK) {
        // Stage A: 128x32 fp32 -> bf16 LDS (float4 global loads)
        for (int u = tid; u < (TM * TK) / 4; u += 256) {
            const int row = u >> 3, c4 = (u & 7) * 4;
            const float4 f =
                *reinterpret_cast<const float4*>(&x[(mBase + row) * K + kc + c4]);
            As[row][c4 + 0] = (__bf16)f.x;
            As[row][c4 + 1] = (__bf16)f.y;
            As[row][c4 + 2] = (__bf16)f.z;
            As[row][c4 + 3] = (__bf16)f.w;
            if (kc + TK < K)  // prefetch next K-chunk (global_prefetch_b8)
                __builtin_prefetch(&x[(mBase + row) * K + kc + TK + c4], 0, 1);
        }
        // Stage B transposed: 32x64 fp32 -> BsT[n][k] bf16
        for (int u = tid; u < (TK * TN) / 4; u += 256) {
            const int row = u >> 4, c4 = (u & 15) * 4;
            const float4 f = *reinterpret_cast<const float4*>(
                &W[(size_t)(kc + row) * N + nBase + c4]);
            BsT[c4 + 0][row] = (__bf16)f.x;
            BsT[c4 + 1][row] = (__bf16)f.y;
            BsT[c4 + 2][row] = (__bf16)f.z;
            BsT[c4 + 3][row] = (__bf16)f.w;
        }
        __syncthreads();

        v16bf af[2], bfr[2];
        for (int i = 0; i < 2; ++i)
            af[i] = loadA(&As[waveM * 32 + i * 16 + lo16][0], half);
        for (int j = 0; j < 2; ++j)
            bfr[j] = loadB(&BsT[waveN * 32 + j * 16 + lo16][0], half);
        for (int i = 0; i < 2; ++i)
            for (int j = 0; j < 2; ++j)
                acc[i][j] = wmma_bf16(af[i], bfr[j], acc[i][j]);
        __syncthreads();
    }

    // Epilogue: C/D layout lane = column (lo16), VGPR r = row r + 8*half.
    for (int i = 0; i < 2; ++i)
        for (int j = 0; j < 2; ++j) {
            const int col = nBase + waveN * 32 + j * 16 + lo16;
            const float bv = bias[col];
            for (int r = 0; r < 8; ++r) {
                const size_t row = mBase + waveM * 32 + i * 16 + r + 8 * half;
                out[row * (size_t)N + col] = (__bf16)(acc[i][j][r] + bv);
            }
        }
}

// ---------------------------------------------------------------------------
// Attention: one wave handles one (b, h, 16-query tile). Window=16 => keys
// live in the current and previous 16-s tiles only (32 keys).
// ---------------------------------------------------------------------------
__global__ __launch_bounds__(128) void attn_kernel(
    const __bf16* __restrict__ qkv, __bf16* __restrict__ aout)
{
    __shared__ __align__(16) __bf16 Psh[4][16][32];    // probs, A-layout staging
    __shared__ __align__(16) __bf16 Vsh[4][128][32];   // V^T: [d][s]

    const int wave = threadIdx.x >> 5, lane = threadIdx.x & 31;
    const int lo16 = lane & 15, half = lane >> 4;
    const int gt = blockIdx.x * 4 + wave;   // 32768 wave-tasks total
    const int qt = gt & 7;                  // query tile (T/16 = 8)
    const int bh = gt >> 3;
    const int h  = bh & 7;
    const int b  = bh >> 3;

    const size_t rowS = (size_t)kN1;
    const __bf16* base = qkv + (size_t)b * kT * rowS + (size_t)h * kD;
    const int sbase0 = qt * 16 - 16;        // first key index (may be < 0)

    // Q fragments: 16 rows x 128 d, as 4 K-chunks of 32.
    v16bf qf[4];
    {
        const int t = qt * 16 + lo16;
        const __bf16* qrow = base + (size_t)t * rowS;
        for (int kcc = 0; kcc < 4; ++kcc)
            qf[kcc] = loadA(qrow + kcc * 32, half);
    }

    // Stage V^T into LDS: lane owns key row s = sbase0 + lane (clamped; rows
    // outside the window get probability 0 from the mask, so garbage is safe).
    {
        int s = sbase0 + lane;
        s = s < 0 ? 0 : s;
        const __bf16* vrow = base + (size_t)s * rowS + 2 * kC;
        for (int c8 = 0; c8 < 16; ++c8) {
            v8bf v = ld8(vrow + c8 * 8);
            for (int j = 0; j < 8; ++j)
                Vsh[wave][c8 * 8 + j][lane] = v[j];
        }
    }

    // Scores: two 16x16 tiles (prev, cur), each K=128 via 4 WMMAs.
    v8f sc[2];
    for (int st = 0; st < 2; ++st) {
        const int s  = sbase0 + st * 16 + lo16;   // lane's key column
        const int sl = s < 0 ? 0 : s;
        const __bf16* krow = base + (size_t)sl * rowS + kC;
        v8f a = {};
        for (int kcc = 0; kcc < 4; ++kcc) {
            v16bf kf = loadB(krow + kcc * 32, half);   // contiguous d-run
            a = wmma_bf16(qf[kcc], kf, a);
        }
        const float scale = 0.08838834764831845f;      // 1/sqrt(128)
        for (int r = 0; r < 8; ++r) {
            const int mr = r + 8 * half;
            const int t  = qt * 16 + mr;
            const bool valid = (s >= 0) && (s <= t) && (t - s <= kWin);
            a[r] = valid ? a[r] * scale : -1e30f;
        }
        sc[st] = a;
    }

    // Row softmax: row values live across the 16 lanes of each half-wave
    // (one column per lane) => shfl_xor reduce over masks 1,2,4,8.
    for (int r = 0; r < 8; ++r) {
        float a = sc[0][r], c = sc[1][r];
        float mx = fmaxf(a, c);
        for (int m2 = 1; m2 <= 8; m2 <<= 1)
            mx = fmaxf(mx, __shfl_xor(mx, m2, 32));
        const float ea = __expf(a - mx), ec = __expf(c - mx);
        float sum = ea + ec;
        for (int m2 = 1; m2 <= 8; m2 <<= 1)
            sum += __shfl_xor(sum, m2, 32);
        const float inv = 1.0f / sum;
        const int mr = r + 8 * half;
        Psh[wave][mr][lo16]      = (__bf16)(ea * inv);
        Psh[wave][mr][16 + lo16] = (__bf16)(ec * inv);
    }

    __syncthreads();   // publish Psh + Vsh (uniform across all 4 waves)

    // P (16x32) x V (32x128): one K=32 WMMA per 16-wide d tile.
    const v16bf pf = loadA(&Psh[wave][lo16][0], half);
    __bf16* orow = aout + (size_t)b * kT * kC + (size_t)h * kD;
    const v8f zero = {};
    for (int nt = 0; nt < 8; ++nt) {
        v16bf vf = loadB(&Vsh[wave][nt * 16 + lo16][0], half);
        v8f o = wmma_bf16(pf, vf, zero);
        const int col = nt * 16 + lo16;
        for (int r = 0; r < 8; ++r) {
            const int t = qt * 16 + r + 8 * half;
            orow[(size_t)t * kC + col] = (__bf16)o[r];
        }
    }
}

// ---------------------------------------------------------------------------
// GEMM 2: out[M=65536, N=1024] = (attn * gate)[M,K=1024] @ W_proj + b_proj
// ---------------------------------------------------------------------------
__global__ __launch_bounds__(256) void proj_gemm_kernel(
    const __bf16* __restrict__ A, const float* __restrict__ gate,
    const float* __restrict__ W, const float* __restrict__ bias,
    float* __restrict__ out)
{
    constexpr int K = kC, N = kC, TM = 128, TN = 64, TK = 32;
    __shared__ __align__(16) __bf16 As[TM][TK];
    __shared__ __align__(16) __bf16 BsT[TN][TK];

    const int tid  = threadIdx.x;
    const int wave = tid >> 5, lane = tid & 31;
    const int lo16 = lane & 15, half = lane >> 4;
    const int waveM = wave >> 1, waveN = wave & 1;
    const size_t mBase = (size_t)blockIdx.y * TM;
    const int    nBase = blockIdx.x * TN;

    v8f acc[2][2] = {};

    for (int kc = 0; kc < K; kc += TK) {
        // Stage A: bf16 loads, fused gate multiply (gate indexes K dim = C)
        for (int u = tid; u < (TM * TK) / 8; u += 256) {
            const int row = u >> 2, c8 = (u & 3) * 8;
            v8bf a = ld8(&A[(mBase + row) * K + kc + c8]);
            for (int j = 0; j < 8; ++j)
                As[row][c8 + j] = (__bf16)((float)a[j] * gate[kc + c8 + j]);
            if (kc + TK < K)
                __builtin_prefetch(&A[(mBase + row) * K + kc + TK + c8], 0, 1);
        }
        for (int u = tid; u < (TK * TN) / 4; u += 256) {
            const int row = u >> 4, c4 = (u & 15) * 4;
            const float4 f = *reinterpret_cast<const float4*>(
                &W[(size_t)(kc + row) * N + nBase + c4]);
            BsT[c4 + 0][row] = (__bf16)f.x;
            BsT[c4 + 1][row] = (__bf16)f.y;
            BsT[c4 + 2][row] = (__bf16)f.z;
            BsT[c4 + 3][row] = (__bf16)f.w;
        }
        __syncthreads();

        v16bf af[2], bfr[2];
        for (int i = 0; i < 2; ++i)
            af[i] = loadA(&As[waveM * 32 + i * 16 + lo16][0], half);
        for (int j = 0; j < 2; ++j)
            bfr[j] = loadB(&BsT[waveN * 32 + j * 16 + lo16][0], half);
        for (int i = 0; i < 2; ++i)
            for (int j = 0; j < 2; ++j)
                acc[i][j] = wmma_bf16(af[i], bfr[j], acc[i][j]);
        __syncthreads();
    }

    for (int i = 0; i < 2; ++i)
        for (int j = 0; j < 2; ++j) {
            const int col = nBase + waveN * 32 + j * 16 + lo16;
            const float bv = bias[col];
            for (int r = 0; r < 8; ++r) {
                const size_t row = mBase + waveM * 32 + i * 16 + r + 8 * half;
                out[row * (size_t)N + col] = acc[i][j][r] + bv;
            }
        }
}

// ---------------------------------------------------------------------------
// Launch
// ---------------------------------------------------------------------------
extern "C" void kernel_launch(void* const* d_in, const int* in_sizes, int n_in,
                              void* d_out, int out_size, void* d_ws, size_t ws_size,
                              hipStream_t stream) {
    const float* x      = (const float*)d_in[0];
    const float* W_qkv  = (const float*)d_in[1];
    const float* b_qkv  = (const float*)d_in[2];
    const float* W_proj = (const float*)d_in[3];
    const float* b_proj = (const float*)d_in[4];
    const float* gate   = (const float*)d_in[5];
    float* out = (float*)d_out;

    // Workspace: qkv bf16 (402.7 MB) then attn-out bf16 (134.2 MB)
    __bf16* qkv_ws  = (__bf16*)d_ws;
    __bf16* attn_ws = qkv_ws + (size_t)kB * kT * kN1;

    // 1) QKV projection: M=65536, N=3072, K=1024
    qkv_gemm_kernel<<<dim3(kN1 / 64, (kB * kT) / 128), 256, 0, stream>>>(
        x, W_qkv, b_qkv, qkv_ws);

    // 2) Attention: B*H*(T/16) = 32768 wave-tasks, 4 waves/block
    attn_kernel<<<dim3((kB * kH * (kT / 16)) / 4), 128, 0, stream>>>(
        qkv_ws, attn_ws);

    // 3) Output projection: M=65536, N=1024, K=1024
    proj_gemm_kernel<<<dim3(kC / 64, (kB * kT) / 128), 256, 0, stream>>>(
        attn_ws, gate, W_proj, b_proj, out);
}
